// VisualContrastAttention_88905823027853
// MI455X (gfx1250) — matrix operand order repaired and checked by reading, not compiled, not run on hardware
//
#include <hip/hip_runtime.h>
#include <hip/hip_bf16.h>
#include <math.h>

// ---------- types ----------
typedef __attribute__((ext_vector_type(16))) __bf16 bf16x16;
typedef __attribute__((ext_vector_type(8)))  float  floatx8;
typedef __attribute__((ext_vector_type(4)))  unsigned int uintx4;

union ABFrag { bf16x16 v; __bf16 e[16]; uintx4 q[2]; };
union Pack8  { __bf16 h[8]; uintx4 q; };

__device__ __forceinline__ floatx8 zero8() {
  floatx8 z;
#pragma unroll
  for (int i = 0; i < 8; ++i) z[i] = 0.f;
  return z;
}

// A fragment: 16xK tile, row-major (K contiguous, stride ld).
// Lane L: row = L&15 ; elems 0..7 -> K = 8g+e ; elems 8..15 -> K = 16+8g+(e-8)
__device__ __forceinline__ bf16x16 load_a(const __bf16* p, int ld) {
  const int lane = threadIdx.x & 31, g = lane >> 4, r = lane & 15;
  const __bf16* a = p + (size_t)r * ld + 8 * g;
  ABFrag f;
  f.q[0] = *reinterpret_cast<const uintx4*>(a);
  f.q[1] = *reinterpret_cast<const uintx4*>(a + 16);
  return f.v;
}

// B fragment: Kx16 tile where B^T is row-major in memory (p -> element [col0][k0],
// stride ld between cols).  Lane L: col = L&15 ; elems 0..7 -> K = 8g+e ; elems 8..15 -> K = 16+8g+(e-8)
__device__ __forceinline__ bf16x16 load_b(const __bf16* p, int ld) {
  const int lane = threadIdx.x & 31, g = lane >> 4, r = lane & 15;
  const __bf16* a = p + (size_t)r * ld + 8 * g;
  ABFrag f;
  f.q[0] = *reinterpret_cast<const uintx4*>(a);
  f.q[1] = *reinterpret_cast<const uintx4*>(a + 16);
  return f.v;
}

__device__ __forceinline__ floatx8 wmma_bf(bf16x16 a, bf16x16 b, floatx8 c) {
  return __builtin_amdgcn_wmma_f32_16x16x32_bf16(false, a, false, b, (short)0, c, false, false);
}

// CDNA5 async global->LDS copy (ASYNCcnt) + wait.  VDST holds the LDS byte
// address (low 32 bits of a generic __shared__ pointer, ISA 10.2).
__device__ __forceinline__ void async_b128(unsigned int lds_addr, const __bf16* g) {
  asm volatile("global_load_async_to_lds_b128 %0, %1, off"
               :: "v"(lds_addr), "v"(g) : "memory");
}
__device__ __forceinline__ void wait_async0() {
  asm volatile("s_wait_asynccnt 0x0" ::: "memory");
}

// ---------- constants ----------
#define NB    32
#define NHEAD 12
#define NTOK  1024
#define HD    64
#define CDIM  768
#define BH    (NB*NHEAD)            // 384
#define QELEM ((size_t)BH*NTOK*HD)  // 25165824

// ---------- fp32 -> bf16 ----------
__global__ void k_f2bf(const float* __restrict__ in, __bf16* __restrict__ out, int n) {
  int i = blockIdx.x * 256 + threadIdx.x;
  if (i < n) out[i] = (__bf16)in[i];
}

// =====================================================================
// Big GEMM core: block = 128 threads (4 waves), tile 128(M) x 64(N),
// wave computes 32x64 (2 m-tiles x 4 n-tiles).  B tile (64x32 bf16, 4KB)
// staged in double-buffered LDS via global_load_async_to_lds_b128;
// A fragments register-double-buffered from global (L2-resident).
// K loop manually unrolled x2 (ping/pong) so no register rotation movs
// and literal LDS buffer offsets.
// =====================================================================
#define GEMM_STAGE_B(BUF, KT)                                                  \
  _Pragma("unroll") for (int u = 0; u < 2; ++u) {                              \
    const int c = tid * 2 + u, row = c >> 2, ko = (c & 3) * 8;                 \
    async_b128((unsigned int)(uintptr_t)&bsmem[BUF][row * 32 + ko],            \
               bptr + (size_t)(n0 + row) * CDIM + (KT) + ko);                  \
  }

#define GEMM_COMPUTE(BUF, AREG, ACC)                                           \
  {                                                                            \
    bf16x16 bfr[4];                                                            \
    _Pragma("unroll") for (int nt = 0; nt < 4; ++nt)                           \
      bfr[nt] = load_b(&bsmem[BUF][(nt * 16) * 32], 32);                       \
    _Pragma("unroll") for (int mi = 0; mi < 2; ++mi)                           \
      _Pragma("unroll") for (int nt = 0; nt < 4; ++nt)                         \
        ACC[mi][nt] = wmma_bf(AREG[mi], bfr[nt], ACC[mi][nt]);                 \
  }

#define GEMM_CORE(A_PTR, B_PTR, ACC)                                           \
  __shared__ __bf16 bsmem[2][64 * 32];                                         \
  const int tid  = threadIdx.x;                                                \
  const int wave = tid >> 5, lane = tid & 31;                                  \
  const int g = lane >> 4, cidx = lane & 15;                                   \
  const int m0 = blockIdx.y * 128 + wave * 32;                                 \
  const int n0 = blockIdx.x * 64;                                              \
  const __bf16* aptr = (A_PTR);                                                \
  const __bf16* bptr = (B_PTR);                                                \
  floatx8 ACC[2][4];                                                           \
  _Pragma("unroll") for (int mi = 0; mi < 2; ++mi)                             \
    _Pragma("unroll") for (int nt = 0; nt < 4; ++nt) ACC[mi][nt] = zero8();    \
  bf16x16 aA[2], aB[2];                                                        \
  /* prologue: stage B(kt=0) into buf0, load A(kt=0) */                        \
  GEMM_STAGE_B(0, 0)                                                           \
  _Pragma("unroll") for (int mi = 0; mi < 2; ++mi)                             \
    aA[mi] = load_a(aptr + (size_t)(m0 + mi * 16) * CDIM, CDIM);               \
  wait_async0();                                                               \
  __syncthreads();                                                             \
  for (int ki = 0; ki < CDIM / 32; ki += 2) {                                  \
    /* ---- step 1: consume buf0/aA, prefetch buf1/aB at kt1 ---- */           \
    const int kt1 = (ki + 1) * 32;            /* always < CDIM */              \
    GEMM_STAGE_B(1, kt1)                                                       \
    _Pragma("unroll") for (int mi = 0; mi < 2; ++mi)                           \
      aB[mi] = load_a(aptr + (size_t)(m0 + mi * 16) * CDIM + kt1, CDIM);       \
    GEMM_COMPUTE(0, aA, ACC)                                                   \
    wait_async0();                                                             \
    __syncthreads();                                                           \
    /* ---- step 2: consume buf1/aB, prefetch buf0/aA at kt2 ---- */           \
    const int kt2 = (ki + 2) * 32;                                             \
    if (kt2 < CDIM) {                                                          \
      GEMM_STAGE_B(0, kt2)                                                     \
      _Pragma("unroll") for (int mi = 0; mi < 2; ++mi)                         \
        aA[mi] = load_a(aptr + (size_t)(m0 + mi * 16) * CDIM + kt2, CDIM);     \
    }                                                                          \
    GEMM_COMPUTE(1, aB, ACC)                                                   \
    wait_async0();                                                             \
    __syncthreads();                                                           \
  }

// ---------- QKV GEMM: [32768,768] x [2304,768]^T, scatter epilogue ----------
__global__ __launch_bounds__(128) void k_qkv_gemm(const __bf16* __restrict__ X,
                                                  const __bf16* __restrict__ W,
                                                  __bf16* __restrict__ qbf,
                                                  __bf16* __restrict__ kbf,
                                                  __bf16* __restrict__ vt) {
  GEMM_CORE(X, W, acc)
#pragma unroll
  for (int nt = 0; nt < 4; ++nt) {
    const int o = n0 + nt * 16 + cidx;
    const int j = o / CDIM, rem = o - j * CDIM;
    const int head = rem >> 6, e = rem & 63;
#pragma unroll
    for (int mi = 0; mi < 2; ++mi)
#pragma unroll
      for (int r = 0; r < 8; ++r) {
        const int mg = m0 + mi * 16 + r + 8 * g;
        const int b = mg >> 10, n = mg & 1023;
        const size_t bh = (size_t)b * NHEAD + head;
        const __bf16 bv = (__bf16)acc[mi][nt][r];
        if (j == 0)      qbf[(bh * NTOK + n) * HD + e] = bv;
        else if (j == 1) kbf[(bh * NTOK + n) * HD + e] = bv;
        else             vt [(bh * HD + e) * NTOK + n] = bv;   // v transposed
      }
  }
}

// ---------- output projection GEMM + bias -> fp32 ----------
__global__ __launch_bounds__(128) void k_proj_gemm(const __bf16* __restrict__ OP,
                                                   const __bf16* __restrict__ W,
                                                   const float* __restrict__ bias,
                                                   float* __restrict__ out) {
  GEMM_CORE(OP, W, acc)
#pragma unroll
  for (int nt = 0; nt < 4; ++nt) {
    const int o = n0 + nt * 16 + cidx;
    const float bz = bias[o];
#pragma unroll
    for (int mi = 0; mi < 2; ++mi)
#pragma unroll
      for (int r = 0; r < 8; ++r) {
        const int mg = m0 + mi * 16 + r + 8 * g;
        out[(size_t)mg * CDIM + o] = acc[mi][nt][r] + bz;
      }
  }
}

// ---------- gq = mean over tokens ----------
__global__ __launch_bounds__(64) void k_gq(const __bf16* __restrict__ qbf, float* __restrict__ gq) {
  const int bh = blockIdx.x, e = threadIdx.x;
  const __bf16* p = qbf + (size_t)bh * NTOK * HD + e;
  float s = 0.f;
  for (int n = 0; n < NTOK; ++n) s += (float)p[(size_t)n * HD];
  gq[bh * HD + e] = s * (1.0f / 1024.0f);
}

// ---------- depthwise 3x3 conv + exact GELU + 4x4 avg pool ----------
__global__ __launch_bounds__(256) void k_convpool(const __bf16* __restrict__ qbf,
                                                  const float* __restrict__ cw,
                                                  float* __restrict__ tbase) {
  const int bh = blockIdx.x;
  const int c = threadIdx.x & 63, mb = threadIdx.x >> 6;
  float w[9];
#pragma unroll
  for (int i = 0; i < 9; ++i) w[i] = cw[c * 9 + i];
  const __bf16* q = qbf + (size_t)bh * NTOK * HD + c;
  for (int mi = 0; mi < 16; ++mi) {
    const int m = mb + 4 * mi;
    const int py = m >> 3, px = m & 7;
    float accp = 0.f;
    for (int dy = 0; dy < 4; ++dy)
      for (int dx = 0; dx < 4; ++dx) {
        const int y = py * 4 + dy, x = px * 4 + dx;
        float s = 0.f;
#pragma unroll
        for (int ky = 0; ky < 3; ++ky) {
          const int yy = y + ky - 1;
          if (yy < 0 || yy > 31) continue;
#pragma unroll
          for (int kx = 0; kx < 3; ++kx) {
            const int xx = x + kx - 1;
            if (xx < 0 || xx > 31) continue;
            s += w[ky * 3 + kx] * (float)q[(size_t)(yy * 32 + xx) * HD];
          }
        }
        accp += 0.5f * s * (1.0f + erff(s * 0.70710678118654752f));
      }
    tbase[((size_t)bh * 64 + m) * HD + c] = accp * (1.0f / 16.0f);
  }
}

// ---------- t_pos / t_neg = t_base + gq @ W{pos,neg}^T  (bf16 out) ----------
__global__ __launch_bounds__(256) void k_temb(const float* __restrict__ tbase,
                                              const float* __restrict__ gq,
                                              const float* __restrict__ Wp,
                                              const float* __restrict__ Wn,
                                              __bf16* __restrict__ tpos,
                                              __bf16* __restrict__ tneg) {
  const int bh = blockIdx.x;
  __shared__ float gs[HD];
  if (threadIdx.x < HD) gs[threadIdx.x] = gq[bh * HD + threadIdx.x];
  __syncthreads();
  const int c = threadIdx.x & 63, mb = threadIdx.x >> 6;
  float pv = 0.f, nv = 0.f;
  for (int f = 0; f < HD; ++f) { pv += gs[f] * Wp[c * HD + f]; nv += gs[f] * Wn[c * HD + f]; }
  for (int mi = 0; mi < 16; ++mi) {
    const int m = mb + 4 * mi;
    const float tb = tbase[((size_t)bh * 64 + m) * HD + c];
    tpos[(size_t)bh * 4096 + m * HD + c] = (__bf16)(tb + pv);
    tneg[(size_t)bh * 4096 + m * HD + c] = (__bf16)(tb + nv);
  }
}

// ---------- stage 1: pooled tokens attend to k/v (flash-style, pos & neg) ----------
__global__ __launch_bounds__(128) void k_stage1(const __bf16* __restrict__ kbf,
                                                const __bf16* __restrict__ vt,
                                                const __bf16* __restrict__ tpos,
                                                const __bf16* __restrict__ tneg,
                                                const float* __restrict__ n1w,
                                                const float* __restrict__ lam1p,
                                                __bf16* __restrict__ vct) {
  const int bh = blockIdx.x;
  const int wave = threadIdx.x >> 5, lane = threadIdx.x & 31;
  const int g = lane >> 4, cidx = lane & 15;
  const float scale = 0.125f;
  const float lam1 = lam1p[0];

  bf16x16 btp[2], btn[2];             // B-frags of T^T for this wave's m-tile
#pragma unroll
  for (int kt = 0; kt < 2; ++kt) {
    btp[kt] = load_b(tpos + (size_t)bh * 4096 + (size_t)(wave * 16) * HD + kt * 32, HD);
    btn[kt] = load_b(tneg + (size_t)bh * 4096 + (size_t)(wave * 16) * HD + kt * 32, HD);
  }
  floatx8 op[4], on[4];
#pragma unroll
  for (int i = 0; i < 4; ++i) { op[i] = zero8(); on[i] = zero8(); }
  float mxp = -1e30f, lp = 0.f, mxn = -1e30f, ln_ = 0.f;

  for (int ch = 0; ch < 16; ++ch) {
    const __bf16* kc = kbf + (size_t)bh * NTOK * HD + (size_t)ch * 64 * HD;
    bf16x16 ka[4][2];
#pragma unroll
    for (int a = 0; a < 4; ++a)
#pragma unroll
      for (int kt = 0; kt < 2; ++kt)
        ka[a][kt] = load_a(kc + (size_t)(a * 16) * HD + kt * 32, HD);

    floatx8 sp[4], sn[4];
#pragma unroll
    for (int a = 0; a < 4; ++a) {
      sp[a] = zero8(); sn[a] = zero8();
#pragma unroll
      for (int kt = 0; kt < 2; ++kt) {
        sp[a] = wmma_bf(ka[a][kt], btp[kt], sp[a]);   // S^T = K_chunk @ T^T
        sn[a] = wmma_bf(ka[a][kt], btn[kt], sn[a]);
      }
    }
    bf16x16 bv[2][4];                   // V B-frags from transposed v
#pragma unroll
    for (int kt = 0; kt < 2; ++kt)
#pragma unroll
      for (int et = 0; et < 4; ++et)
        bv[kt][et] = load_b(vt + (size_t)bh * HD * NTOK + (size_t)(et * 16) * NTOK + ch * 64 + kt * 32, NTOK);

    { // ---- pos online-softmax update ----
      float cm = -1e30f;
#pragma unroll
      for (int a = 0; a < 4; ++a)
#pragma unroll
        for (int r = 0; r < 8; ++r) cm = fmaxf(cm, sp[a][r]);
      cm *= scale;
      cm = fmaxf(cm, __shfl_xor(cm, 16, 32));
      const float mnew = fmaxf(mxp, cm);
      const float alpha = __expf(mxp - mnew);
      lp *= alpha;
#pragma unroll
      for (int et = 0; et < 4; ++et)
#pragma unroll
        for (int r = 0; r < 8; ++r) op[et][r] *= alpha;
      ABFrag pa[2]; float rs = 0.f;
#pragma unroll
      for (int kt = 0; kt < 2; ++kt)
#pragma unroll
        for (int h2 = 0; h2 < 2; ++h2)
#pragma unroll
          for (int r = 0; r < 8; ++r) {
            const float pe = __expf(sp[2 * kt + h2][r] * scale - mnew);
            rs += pe;
            pa[kt].e[h2 * 8 + r] = (__bf16)pe;     // C-layout of S^T == A-layout of P
          }
      rs += __shfl_xor(rs, 16, 32);
      lp += rs; mxp = mnew;
#pragma unroll
      for (int kt = 0; kt < 2; ++kt)
#pragma unroll
        for (int et = 0; et < 4; ++et) op[et] = wmma_bf(pa[kt].v, bv[kt][et], op[et]);
    }
    { // ---- neg online-softmax update ----
      float cm = -1e30f;
#pragma unroll
      for (int a = 0; a < 4; ++a)
#pragma unroll
        for (int r = 0; r < 8; ++r) cm = fmaxf(cm, sn[a][r]);
      cm *= scale;
      cm = fmaxf(cm, __shfl_xor(cm, 16, 32));
      const float mnew = fmaxf(mxn, cm);
      const float alpha = __expf(mxn - mnew);
      ln_ *= alpha;
#pragma unroll
      for (int et = 0; et < 4; ++et)
#pragma unroll
        for (int r = 0; r < 8; ++r) on[et][r] *= alpha;
      ABFrag pa[2]; float rs = 0.f;
#pragma unroll
      for (int kt = 0; kt < 2; ++kt)
#pragma unroll
        for (int h2 = 0; h2 < 2; ++h2)
#pragma unroll
          for (int r = 0; r < 8; ++r) {
            const float pe = __expf(sn[2 * kt + h2][r] * scale - mnew);
            rs += pe;
            pa[kt].e[h2 * 8 + r] = (__bf16)pe;
          }
      rs += __shfl_xor(rs, 16, 32);
      ln_ += rs; mxn = mnew;
#pragma unroll
      for (int kt = 0; kt < 2; ++kt)
#pragma unroll
        for (int et = 0; et < 4; ++et) on[et] = wmma_bf(pa[kt].v, bv[kt][et], on[et]);
    }
  }

  // finalize: v_hat = O/l, contrast, rmsnorm over e, store transposed vct[e][m]
  float linp[8], linn[8];
#pragma unroll
  for (int r = 0; r < 8; ++r) {
    linp[r] = 1.0f / __shfl(lp,  r + 8 * g, 32);
    linn[r] = 1.0f / __shfl(ln_, r + 8 * g, 32);
  }
  float dd[4][8], ss[8];
#pragma unroll
  for (int r = 0; r < 8; ++r) ss[r] = 0.f;
#pragma unroll
  for (int et = 0; et < 4; ++et)
#pragma unroll
    for (int r = 0; r < 8; ++r) {
      const float d = op[et][r] * linp[r] - lam1 * on[et][r] * linn[r];
      dd[et][r] = d; ss[r] += d * d;
    }
#pragma unroll
  for (int r = 0; r < 8; ++r) {
    ss[r] += __shfl_xor(ss[r], 1, 32);
    ss[r] += __shfl_xor(ss[r], 2, 32);
    ss[r] += __shfl_xor(ss[r], 4, 32);
    ss[r] += __shfl_xor(ss[r], 8, 32);
  }
  float sc[8];
#pragma unroll
  for (int r = 0; r < 8; ++r) sc[r] = 0.5f / (sqrtf(ss[r]) * 0.125f + 1e-6f);
#pragma unroll
  for (int et = 0; et < 4; ++et) {
    const int e = et * 16 + cidx;
    const float nw = n1w[e];
    Pack8 pk;
#pragma unroll
    for (int r = 0; r < 8; ++r) pk.h[r] = (__bf16)(dd[et][r] * sc[r] * nw);
    *reinterpret_cast<uintx4*>(vct + ((size_t)bh * 64 + e) * 64 + wave * 16 + 8 * g) = pk.q;
  }
}

// ---------- stage 2: q attends to contrast tokens ----------
__global__ __launch_bounds__(128) void k_stage2(const __bf16* __restrict__ qbf,
                                                const __bf16* __restrict__ tpos,
                                                const __bf16* __restrict__ tneg,
                                                const __bf16* __restrict__ vct,
                                                const float* __restrict__ n2w,
                                                const float* __restrict__ lam2p,
                                                __bf16* __restrict__ opbf) {
  const int ch = blockIdx.x, bh = blockIdx.y;
  const int wave = threadIdx.x >> 5, lane = threadIdx.x & 31;
  const int g = lane >> 4, cidx = lane & 15;
  const float scale = 0.125f;
  const float lam2 = lam2p[0];

  bf16x16 bq[2];                    // B-frags of q^T for this wave's n-tile
#pragma unroll
  for (int kt = 0; kt < 2; ++kt)
    bq[kt] = load_b(qbf + (size_t)bh * NTOK * HD + (size_t)(ch * 64 + wave * 16) * HD + kt * 32, HD);

  floatx8 sp[4], sn[4];
#pragma unroll
  for (int a = 0; a < 4; ++a) {
    sp[a] = zero8(); sn[a] = zero8();
#pragma unroll
    for (int kt = 0; kt < 2; ++kt) {
      bf16x16 atp = load_a(tpos + (size_t)bh * 4096 + (size_t)(a * 16) * HD + kt * 32, HD);
      bf16x16 atn = load_a(tneg + (size_t)bh * 4096 + (size_t)(a * 16) * HD + kt * 32, HD);
      sp[a] = wmma_bf(atp, bq[kt], sp[a]);        // S2^T = T @ q^T
      sn[a] = wmma_bf(atn, bq[kt], sn[a]);
    }
  }
  float mxp = -1e30f, mxn = -1e30f;
#pragma unroll
  for (int a = 0; a < 4; ++a)
#pragma unroll
    for (int r = 0; r < 8; ++r) { mxp = fmaxf(mxp, sp[a][r]); mxn = fmaxf(mxn, sn[a][r]); }
  mxp *= scale; mxn *= scale;
  mxp = fmaxf(mxp, __shfl_xor(mxp, 16, 32));
  mxn = fmaxf(mxn, __shfl_xor(mxn, 16, 32));
  float sump = 0.f, sumn = 0.f;
#pragma unroll
  for (int a = 0; a < 4; ++a)
#pragma unroll
    for (int r = 0; r < 8; ++r) {
      float pe = __expf(sp[a][r] * scale - mxp); sp[a][r] = pe; sump += pe;
      float ne = __expf(sn[a][r] * scale - mxn); sn[a][r] = ne; sumn += ne;
    }
  sump += __shfl_xor(sump, 16, 32);
  sumn += __shfl_xor(sumn, 16, 32);
  const float ip = 1.0f / sump, inn = 1.0f / sumn;

  ABFrag pa[2];                     // combined P2 in A-layout
#pragma unroll
  for (int kt = 0; kt < 2; ++kt)
#pragma unroll
    for (int h2 = 0; h2 < 2; ++h2)
#pragma unroll
      for (int r = 0; r < 8; ++r)
        pa[kt].e[h2 * 8 + r] = (__bf16)(sp[2 * kt + h2][r] * ip - lam2 * sn[2 * kt + h2][r] * inn);

  floatx8 oa[4];
#pragma unroll
  for (int i = 0; i < 4; ++i) oa[i] = zero8();
#pragma unroll
  for (int kt = 0; kt < 2; ++kt)
#pragma unroll
    for (int et = 0; et < 4; ++et) {
      bf16x16 bvc = load_b(vct + (size_t)bh * 4096 + (size_t)(et * 16) * 64 + kt * 32, 64);
      oa[et] = wmma_bf(pa[kt].v, bvc, oa[et]);
    }

  float ss[8];
#pragma unroll
  for (int r = 0; r < 8; ++r) ss[r] = 0.f;
#pragma unroll
  for (int et = 0; et < 4; ++et)
#pragma unroll
    for (int r = 0; r < 8; ++r) ss[r] += oa[et][r] * oa[et][r];
#pragma unroll
  for (int r = 0; r < 8; ++r) {
    ss[r] += __shfl_xor(ss[r], 1, 32);
    ss[r] += __shfl_xor(ss[r], 2, 32);
    ss[r] += __shfl_xor(ss[r], 4, 32);
    ss[r] += __shfl_xor(ss[r], 8, 32);
  }
  float sc[8];
#pragma unroll
  for (int r = 0; r < 8; ++r) sc[r] = 0.5f / (sqrtf(ss[r]) * 0.125f + 1e-6f);

  const int b = bh / NHEAD, head = bh % NHEAD;
#pragma unroll
  for (int et = 0; et < 4; ++et) {
    const int e = et * 16 + cidx;
    const float nw = n2w[e];
#pragma unroll
    for (int r = 0; r < 8; ++r) {
      const int ntok = ch * 64 + wave * 16 + r + 8 * g;
      opbf[((size_t)b * NTOK + ntok) * CDIM + head * HD + e] = (__bf16)(oa[et][r] * sc[r] * nw);
    }
  }
}

// ---------- host launcher ----------
extern "C" void kernel_launch(void* const* d_in, const int* in_sizes, int n_in,
                              void* d_out, int out_size, void* d_ws, size_t ws_size,
                              hipStream_t stream) {
  const float* x     = (const float*)d_in[0];
  const float* Wqkv  = (const float*)d_in[1];
  const float* Wproj = (const float*)d_in[2];
  const float* bproj = (const float*)d_in[3];
  const float* Wpos  = (const float*)d_in[4];
  const float* Wneg  = (const float*)d_in[5];
  const float* convw = (const float*)d_in[6];
  const float* lam1  = (const float*)d_in[7];
  const float* lam2  = (const float*)d_in[8];
  const float* n1w   = (const float*)d_in[9];
  const float* n2w   = (const float*)d_in[10];
  float* out = (float*)d_out;

  unsigned char* wptr = (unsigned char*)d_ws;
  auto take = [&](size_t bytes) -> void* {
    void* r = (void*)wptr;
    wptr += (bytes + 255) & ~(size_t)255;
    return r;
  };
  const size_t XN = (size_t)NB * NTOK * CDIM;      // 25165824
  __bf16* xbf   = (__bf16*)take(XN * 2);
  __bf16* wqb   = (__bf16*)take((size_t)3 * CDIM * CDIM * 2);
  __bf16* wpb   = (__bf16*)take((size_t)CDIM * CDIM * 2);
  __bf16* qbf   = (__bf16*)take(QELEM * 2);
  __bf16* kbf   = (__bf16*)take(QELEM * 2);
  __bf16* vt    = (__bf16*)take(QELEM * 2);
  float*  gq    = (float*) take((size_t)BH * HD * 4);
  float*  tbase = (float*) take((size_t)BH * 4096 * 4);
  __bf16* tpos  = (__bf16*)take((size_t)BH * 4096 * 2);
  __bf16* tneg  = (__bf16*)take((size_t)BH * 4096 * 2);
  __bf16* vct   = (__bf16*)take((size_t)BH * 4096 * 2);
  __bf16* opbf  = (__bf16*)take(XN * 2);

  k_f2bf<<<(unsigned)((XN + 255) / 256), 256, 0, stream>>>(x, xbf, (int)XN);
  k_f2bf<<<(3 * CDIM * CDIM + 255) / 256, 256, 0, stream>>>(Wqkv, wqb, 3 * CDIM * CDIM);
  k_f2bf<<<(CDIM * CDIM + 255) / 256, 256, 0, stream>>>(Wproj, wpb, CDIM * CDIM);

  k_qkv_gemm<<<dim3(36, 256), 128, 0, stream>>>(xbf, wqb, qbf, kbf, vt);
  k_gq<<<BH, 64, 0, stream>>>(qbf, gq);
  k_convpool<<<BH, 256, 0, stream>>>(qbf, convw, tbase);
  k_temb<<<BH, 256, 0, stream>>>(tbase, gq, Wpos, Wneg, tpos, tneg);
  k_stage1<<<BH, 128, 0, stream>>>(kbf, vt, tpos, tneg, n1w, lam1, vct);
  k_stage2<<<dim3(16, BH), 128, 0, stream>>>(qbf, tpos, tneg, vct, n2w, lam2, opbf);
  k_proj_gemm<<<dim3(12, 256), 128, 0, stream>>>(opbf, wpb, bproj, out);

  (void)in_sizes; (void)n_in; (void)out_size; (void)ws_size;
}
